// BKT_RNN_23081154249236
// MI455X (gfx1250) — matrix operand order repaired and checked by reading, not compiled, not run on hardware
//
#include <hip/hip_runtime.h>
#include <hip/hip_bf16.h>
#include <math.h>

#define HID   128
#define G4    512          // 4*HID
#define NROW  (G4 + 16)    // + post_W tile rows (4 real, 12 zero)
#define TLEN  512
#define BATCH 1024
#define MT    16           // batch rows per workgroup
#define NWG   (BATCH / MT) // 64 workgroups
#define NTHR  256          // 8 waves
#define WSDW  68           // LDS row stride in dwords: 16B-aligned frags, conflict-free per 16-lane phase

typedef __attribute__((ext_vector_type(16))) __bf16 v16bf;
typedef __attribute__((ext_vector_type(8)))  float  v8f;
typedef __attribute__((ext_vector_type(8)))  int    v8i;
typedef __attribute__((ext_vector_type(4)))  int    v4i;

union Frag { v8i i; v16bf b; };
union PackBF { __bf16 h[2]; unsigned int u; };

// two 16B-aligned ds_load_b128 per fragment
__device__ __forceinline__ void ld_frag(Frag& f, const unsigned int* p, int second_off) {
    v4i a = *(const v4i*)(p);
    v4i c = *(const v4i*)(p + second_off);
    f.i = __builtin_shufflevector(a, c, 0, 1, 2, 3, 4, 5, 6, 7);
}

__device__ __forceinline__ float sigm(float v) {
    return __builtin_amdgcn_rcpf(1.0f + __expf(-v));
}
__device__ __forceinline__ float tanh_fast(float v) {
    return 1.0f - 2.0f * __builtin_amdgcn_rcpf(__expf(2.0f * v) + 1.0f);
}

__global__ __launch_bounds__(NTHR) void bkt_lstm_fused(
    const float* __restrict__ x,       const float* __restrict__ y,
    const float* __restrict__ W_ih,    const float* __restrict__ W_hh,
    const float* __restrict__ b_ih,    const float* __restrict__ b_hh,
    const float* __restrict__ prior_W, const float* __restrict__ prior_b,
    const float* __restrict__ post_W,  const float* __restrict__ post_b,
    float* __restrict__ out,           float* __restrict__ ws)
{
    // bf16 weights: rows 0..511 = W_hh, rows 512..515 = post_W, 516..527 = 0
    __shared__ __align__(16) unsigned int Wlds[NROW * WSDW];
    __shared__ __align__(16) unsigned int Hlds[2 * MT * WSDW];   // h ping-pong
    __shared__ float paramsLds[MT * 4];
    __shared__ __align__(16) float xbuf[2][MT];
    __shared__ float lossred[MT];

    const int tid  = threadIdx.x;
    const int lane = tid & 31;
    const int wv   = tid >> 5;            // wave 0..7 -> hidden units [wv*16, wv*16+16)
    const int b0   = blockIdx.x * MT;
    const int lo16 = lane & 15;
    const int hiH  = (lane < 16) ? 0 : 1;

    // ---- init: W_hh -> bf16 LDS (dword-packed) ----
    for (int idx = tid; idx < G4 * (HID / 2); idx += NTHR) {
        int n  = idx >> 6;
        int kd = idx & 63;
        PackBF p;
        p.h[0] = (__bf16)W_hh[n * HID + 2 * kd + 0];
        p.h[1] = (__bf16)W_hh[n * HID + 2 * kd + 1];
        Wlds[n * WSDW + kd] = p.u;
    }
    // post_W rows (zero-padded to 16)
    for (int idx = tid; idx < 16 * (HID / 2); idx += NTHR) {
        int p  = idx >> 6;
        int kd = idx & 63;
        PackBF pk;
        if (p < 4) {
            pk.h[0] = (__bf16)post_W[p * HID + 2 * kd + 0];
            pk.h[1] = (__bf16)post_W[p * HID + 2 * kd + 1];
        } else pk.u = 0u;
        Wlds[(G4 + p) * WSDW + kd] = pk.u;
    }
    for (int idx = tid; idx < 2 * MT * WSDW; idx += NTHR) Hlds[idx] = 0u;

    // per-lane constants
    float biasq[4], wihq[4];
#pragma unroll
    for (int q = 0; q < 4; ++q) {
        int n = q * HID + wv * 16 + lo16;
        biasq[q] = b_ih[n] + b_hh[n];
        wihq[q]  = W_ih[n];
    }
    const float pb = (lo16 < 4) ? post_b[lo16] : 0.0f;

    float cst[8];
#pragma unroll
    for (int v = 0; v < 8; ++v) cst[v] = 0.0f;

    float lat = 0.0f, lossacc = 0.0f;
    if (tid < MT) {
        lat = sigm(x[b0 + tid] * prior_W[0] + prior_b[0]);
        xbuf[0][tid] = x[b0 + tid];               // stage x for t=0
    }
    __syncthreads();

    // wave 0: hoist loop-invariant post_W B-fragments into registers
    Frag PB[4];
    if (wv == 0) {
#pragma unroll
        for (int kki = 0; kki < 4; ++kki)
            ld_frag(PB[kki], Wlds + (G4 + lo16) * WSDW + kki * 16 + hiH * 4, 8);
    }

    Frag A[4];

    for (int t = 0; t < TLEN; ++t) {
        const unsigned int* hrd = Hlds + (t & 1) * (MT * WSDW);        // h_{t-1}
        unsigned int*       hwr = Hlds + ((t + 1) & 1) * (MT * WSDW);  // h_t

        // ---- A fragments from h_{t-1} (wave 0 reuses post-proj frags from t-1)
        if (wv != 0 || t == 0) {
            const unsigned int* hrow = hrd + lo16 * WSDW + hiH * 4;
#pragma unroll
            for (int kki = 0; kki < 4; ++kki)
                ld_frag(A[kki], hrow + kki * 16, 8);
        }

        // prefetch x for next step, y for this step (latency hidden under GEMM)
        float ytv = 0.0f;
        if (tid < MT) {
            int tn = (t + 1 < TLEN) ? t + 1 : t;
            xbuf[(t + 1) & 1][tid] = x[tn * BATCH + b0 + tid];
            ytv = y[t * BATCH + b0 + tid];
        }

        // ---- accumulators: x_t*W_ih + (b_ih + b_hh)
        v8f acc[4];
#pragma unroll
        for (int q = 0; q < 4; ++q) {
#pragma unroll
            for (int v = 0; v < 8; ++v)
                acc[q][v] = xbuf[t & 1][v + 8 * hiH] * wihq[q] + biasq[q];
        }

        // ---- gates += h_{t-1} @ W_hh.T  (bf16 WMMA, K=128 in 4x32)
#pragma unroll
        for (int kki = 0; kki < 4; ++kki) {
#pragma unroll
            for (int q = 0; q < 4; ++q) {
                Frag Bf;
                const int n = q * HID + wv * 16 + lo16;
                ld_frag(Bf, Wlds + n * WSDW + kki * 16 + hiH * 4, 8);
                acc[q] = __builtin_amdgcn_wmma_f32_16x16x32_bf16(
                    false, A[kki].b, false, Bf.b, (short)0, acc[q], false, false);
            }
        }

        // ---- LSTM cell in registers -> h_t (bf16) into write buffer
        {
            __bf16* Hw = (__bf16*)hwr;
            const int j = wv * 16 + lo16;
#pragma unroll
            for (int v = 0; v < 8; ++v) {
                int m = v + 8 * hiH;
                float ig = sigm(acc[0][v]);
                float fg = sigm(acc[1][v]);
                float gg = tanh_fast(acc[2][v]);
                float og = sigm(acc[3][v]);
                float c  = fg * cst[v] + ig * gg;
                cst[v]   = c;
                float h  = og * tanh_fast(c);
                Hw[m * (2 * WSDW) + j] = (__bf16)h;
            }
        }
        __syncthreads();   // single barrier per step: h_t (and staged x) visible

        // ---- wave 0: post projection as one WMMA tile + BKT step
        if (wv == 0) {
            const unsigned int* hrow = hwr + lo16 * WSDW + hiH * 4;    // h_t
#pragma unroll
            for (int kki = 0; kki < 4; ++kki)
                ld_frag(A[kki], hrow + kki * 16, 8);

            v8f pacc;
#pragma unroll
            for (int v = 0; v < 8; ++v) pacc[v] = pb;
#pragma unroll
            for (int kki = 0; kki < 4; ++kki)
                pacc = __builtin_amdgcn_wmma_f32_16x16x32_bf16(
                    false, A[kki].b, false, PB[kki].b, (short)0, pacc, false, false);

            if (lo16 < 4) {
#pragma unroll
                for (int v = 0; v < 8; ++v)
                    paramsLds[(v + 8 * hiH) * 4 + lo16] = sigm(pacc[v]);
            }
            // same-wave LDS ordering: stores above precede loads below
            if (tid < MT) {
                float l = paramsLds[tid * 4 + 0];
                float f = paramsLds[tid * 4 + 1];
                float g = paramsLds[tid * 4 + 2];
                float s = paramsLds[tid * 4 + 3];
                float correct = lat * (1.0f - s) + (1.0f - lat) * g;
                float k1 = lat * (1.0f - s) * __builtin_amdgcn_rcpf(correct);
                float k0 = lat * s * __builtin_amdgcn_rcpf(lat * s + (1.0f - lat) * (1.0f - g));
                float mt = k1 * correct + k0 * (1.0f - correct);
                lat = mt * (1.0f - f) + (1.0f - mt) * l;
                out[t * BATCH + b0 + tid]                = correct;
                out[TLEN * BATCH + t * BATCH + b0 + tid] = lat;
                lossacc -= ytv * fmaxf(__logf(correct), -100.0f)
                         + (1.0f - ytv) * fmaxf(__logf(1.0f - correct), -100.0f);
            }
        }
    }

    // ---- deterministic per-WG loss partial
    if (tid < MT) lossred[tid] = lossacc;
    __syncthreads();
    if (tid == 0) {
        float s = 0.0f;
#pragma unroll
        for (int i = 0; i < MT; ++i) s += lossred[i];
        ws[blockIdx.x] = s;
    }
}

__global__ void loss_reduce(const float* __restrict__ ws, float* __restrict__ out) {
    if (threadIdx.x == 0 && blockIdx.x == 0) {
        float s = 0.0f;
        for (int i = 0; i < NWG; ++i) s += ws[i];
        out[2 * TLEN * BATCH] = s / (float)(TLEN * BATCH);
    }
}

extern "C" void kernel_launch(void* const* d_in, const int* in_sizes, int n_in,
                              void* d_out, int out_size, void* d_ws, size_t ws_size,
                              hipStream_t stream) {
    (void)in_sizes; (void)n_in; (void)out_size; (void)ws_size;
    const float* x       = (const float*)d_in[0];
    const float* y       = (const float*)d_in[1];
    const float* W_ih    = (const float*)d_in[2];
    const float* W_hh    = (const float*)d_in[3];
    const float* b_ih    = (const float*)d_in[4];
    const float* b_hh    = (const float*)d_in[5];
    const float* prior_W = (const float*)d_in[6];
    const float* prior_b = (const float*)d_in[7];
    const float* post_W  = (const float*)d_in[8];
    const float* post_b  = (const float*)d_in[9];
    float* out = (float*)d_out;
    float* ws  = (float*)d_ws;

    bkt_lstm_fused<<<NWG, NTHR, 0, stream>>>(x, y, W_ih, W_hh, b_ih, b_hh,
                                             prior_W, prior_b, post_W, post_b,
                                             out, ws);
    loss_reduce<<<1, 32, 0, stream>>>(ws, out);
}
